// GMABSE3_18743237279829
// MI455X (gfx1250) — compile-verified, matrix operand order checked by reading, not run on hardware
//
#include <hip/hip_runtime.h>
#include <hip/hip_bf16.h>
#include <math.h>

// ---------------- problem constants ----------------
#define Hh   8
#define Bb   4
#define Ll   8192
#define Nn   (Bb*Ll)          // 32768 tokens
#define BH   (Bb*Hh)          // 32
#define BHL  (BH*Ll)          // 262144 rows
#define Mm   128              // random features
#define DK   32               // per-head dim
#define CE   48               // value cols: 8 (v0) + 24 (v1) + 1 (ones) + 15 pad
#define LP   132              // padded LDS row length (bank-conflict avoidance)
#define EPSc 1e-4f
#define DNc     0.42044820762685725f   // 32^-0.25
#define DIAGc   0.08838834764831845f   // 0.5 * 32^-0.5
#define RATIOc  0.08838834764831845f   // 128^-0.5

typedef __attribute__((ext_vector_type(16))) __bf16 v16bf;
typedef __attribute__((ext_vector_type(8)))  float  v8f;

// native convert (v_cvt bf16 path on gfx1250)
__device__ __forceinline__ __bf16 f2bf(float f) { return (__bf16)f; }

union bfpair { unsigned u; __bf16 b[2]; };

// K-base held by dword v (0..7) of a 16-bit A/B fragment for this lane-half
// (ISA 7.12.2: VGPR0..3 -> K {0..7}+8*half, VGPR4..7 -> K {16..23}+8*half)
__device__ __forceinline__ int kbase(int v, int half) {
  return (v < 4) ? (2*v + 8*half) : (16 + 8*half + 2*(v-4));
}
__device__ __forceinline__ int kpat(int j, int half) { return kbase(j >> 1, half) + (j & 1); }

// k-feature element k (0..31) of row (n,h):  k0[n,8h..8h+7], then k1[n,8h+c,e]
__device__ __forceinline__ float featld(const float* __restrict__ k0,
                                        const float* __restrict__ k1,
                                        long n, int h, int k) {
  if (k < 8) return k0[n*64 + h*8 + k];
  int j = k - 8;
  return k1[(n*64 + h*8 + j/3)*3 + (j%3)];
}

__device__ __forceinline__ v8f wmma_bf16(v16bf a, v16bf b, v8f c) {
  return __builtin_amdgcn_wmma_f32_16x16x32_bf16(false, a, false, b, (short)0, c, false, false);
}

// Build B fragments of w^T (8 column tiles of 16) from bf16 LDS copy of w.
// B[k][m] = w[m][k]; K-pairs are contiguous -> packed 32-bit LDS reads.
__device__ __forceinline__ void buildWB(const __bf16* swb, int lm, int half, v16bf* wB) {
#pragma unroll
  for (int t = 0; t < 8; ++t)
#pragma unroll
    for (int v = 0; v < 8; ++v) {
      bfpair p; p.u = *(const unsigned*)&swb[(t*16 + lm)*DK + kbase(v, half)];
      wB[t][2*v]   = p.b[0];
      wB[t][2*v+1] = p.b[1];
    }
}

// ---------------- kernel 1: dd stats (row max, diag) -------------------------
__global__ __launch_bounds__(256)
void k_stats(const float* __restrict__ k0, const float* __restrict__ k1,
             const float* __restrict__ w,
             float* __restrict__ diag, float* __restrict__ rowmax) {
  __shared__ __bf16 swb[Mm*DK];
  int tid = threadIdx.x;
  for (int i = tid; i < Mm*DK; i += 256) swb[i] = f2bf(w[i]);
  __syncthreads();
  int lane = tid & 31, wv = tid >> 5, lm = lane & 15, half = lane >> 4;
  v16bf wB[8]; buildWB(swb, lm, half, wB);

  int rowBase = blockIdx.x * 128 + wv * 16;     // rows in (b,h)-major order
  int row = rowBase + lm;                        // this lane's A row
  int bh = row / Ll, l = row % Ll;
  int b = bh >> 3, h = bh & 7;
  long n = (long)b * Ll + l;

  float sq = 0.f;
  v16bf a;
#pragma unroll
  for (int j = 0; j < 16; ++j) {
    float f = featld(k0, k1, n, h, kpat(j, half));
    sq += f * f;
    a[j] = f2bf(f * DNc);
  }
  float cmax[8];
#pragma unroll
  for (int r = 0; r < 8; ++r) cmax[r] = -INFINITY;
  v8f zero = {};
#pragma unroll
  for (int t = 0; t < 8; ++t) {
    v8f acc = wmma_bf16(a, wB[t], zero);
#pragma unroll
    for (int r = 0; r < 8; ++r) cmax[r] = fmaxf(cmax[r], acc[r]);
  }
  // reduce over the 16 lanes of this half (cols) -> rowmax of row r+8*half
#pragma unroll
  for (int r = 0; r < 8; ++r) {
#pragma unroll
    for (int msk = 1; msk < 16; msk <<= 1)
      cmax[r] = fmaxf(cmax[r], __shfl_xor(cmax[r], msk, 32));
  }
  if (lm == 0) {
#pragma unroll
    for (int r = 0; r < 8; ++r) rowmax[rowBase + r + 8*half] = cmax[r];
  }
  float full = sq + __shfl_xor(sq, 16, 32);      // combine the two K-halves
  if (lane < 16) diag[rowBase + lane] = full * DIAGc;
}

// ---------------- kernel 2: global max of dd ---------------------------------
__global__ __launch_bounds__(1024)
void k_gmax(const float* __restrict__ rowmax, float* __restrict__ gmax) {
  __shared__ float s[1024];
  float m = -INFINITY;
  for (int i = threadIdx.x; i < BHL; i += 1024) m = fmaxf(m, rowmax[i]);
  s[threadIdx.x] = m; __syncthreads();
  for (int off = 512; off; off >>= 1) {
    if ((int)threadIdx.x < off) s[threadIdx.x] = fmaxf(s[threadIdx.x], s[threadIdx.x + off]);
    __syncthreads();
  }
  if (threadIdx.x == 0) gmax[0] = s[0];
}

// ---------------- kernel 0: zero ctx -----------------------------------------
__global__ void k_init(float* __restrict__ ctx) {
  int i = blockIdx.x * 256 + threadIdx.x;
  if (i < BH*Mm*CE) ctx[i] = 0.f;
}

// ---------------- kernel 3: ctx = Kp^T @ [V | 1]  (+Ksum in col 32) ----------
__global__ __launch_bounds__(256)
void k_ctx(const float* __restrict__ k0, const float* __restrict__ k1,
           const float* __restrict__ v0, const float* __restrict__ v1,
           const float* __restrict__ w,  const float* __restrict__ diag,
           const float* __restrict__ gmaxp, float* __restrict__ ctx) {
  __shared__ __bf16 swb[Mm*DK];        //  8 KB  w in bf16
  __shared__ __bf16 KpT[Mm*LP];        // 33 KB  [m][l_local] (padded rows)
  __shared__ __bf16 VlT[CE*LP];        // 12.4KB [col][l_local]
  int tid = threadIdx.x;
  for (int i = tid; i < Mm*DK; i += 256) swb[i] = f2bf(w[i]);
  int lane = tid & 31, wv = tid >> 5, lm = lane & 15, half = lane >> 4;
  __syncthreads();
  v16bf wB[8]; buildWB(swb, lm, half, wB);

  int bh = blockIdx.x >> 3, sp = blockIdx.x & 7;   // 8 blocks per (b,h)
  int b = bh >> 3, h = bh & 7;
  float gm = gmaxp[0];
  v8f cacc[3] = {};

  for (int chunk = 0; chunk < 8; ++chunk) {        // 128 l-rows per chunk
    int l0 = sp * 1024 + chunk * 128;
    __syncthreads();                               // LDS reuse guard
    // prefetch next chunk's value rows while this chunk computes
    if (chunk < 7) {
      long np = (long)b * Ll + l0 + 128 + (tid & 127);
      __builtin_prefetch(&v0[np*64 + h*8], 0, 3);
      __builtin_prefetch(&v1[(np*64 + h*8)*3], 0, 3);
    }
    // stage [V | ones | pad] chunk, transposed to [col][l] (col fastest in i
    // for coalesced global reads)
    for (int i = tid; i < 128*CE; i += 256) {
      int li = i / CE, col = i % CE;
      long n = (long)b * Ll + l0 + li;
      float v;
      if (col < 8)        v = v0[n*64 + h*8 + col];
      else if (col < 32)  { int idx = col - 8; v = v1[(n*64 + h*8 + idx/3)*3 + idx%3]; }
      else                v = (col == 32) ? 1.0f : 0.0f;
      VlT[col*LP + li] = f2bf(v);
    }
    // recompute dd for this wave's 16 l-rows, exp -> Kp, store transposed
    {
      int lrow = wv*16 + lm;
      long n = (long)b * Ll + l0 + lrow;
      v16bf a;
#pragma unroll
      for (int j = 0; j < 16; ++j)
        a[j] = f2bf(featld(k0, k1, n, h, kpat(j, half)) * DNc);
      float dg[8];
#pragma unroll
      for (int r = 0; r < 8; ++r)
        dg[r] = diag[bh*Ll + l0 + wv*16 + r + 8*half];
      v8f zero = {};
#pragma unroll
      for (int t = 0; t < 8; ++t) {
        v8f acc = wmma_bf16(a, wB[t], zero);
        int ll0 = wv*16 + 8*half;
#pragma unroll
        for (int r = 0; r < 8; r += 2) {          // packed 32-bit Kp stores
          bfpair p;
          p.b[0] = f2bf(RATIOc * (__expf(acc[r]   - dg[r]   - gm) + EPSc));
          p.b[1] = f2bf(RATIOc * (__expf(acc[r+1] - dg[r+1] - gm) + EPSc));
          *(unsigned*)&KpT[(t*16 + lm)*LP + ll0 + r] = p.u;
        }
      }
    }
    __syncthreads();
    // accumulate ctx tile: this wave owns m-tile = wv, all 3 col-tiles
#pragma unroll
    for (int ks = 0; ks < 4; ++ks) {
      v16bf ka;
#pragma unroll
      for (int v = 0; v < 8; ++v) {
        bfpair p; p.u = *(const unsigned*)&KpT[(wv*16 + lm)*LP + ks*32 + kbase(v, half)];
        ka[2*v] = p.b[0]; ka[2*v+1] = p.b[1];
      }
#pragma unroll
      for (int t = 0; t < 3; ++t) {
        v16bf vb;
#pragma unroll
        for (int v = 0; v < 8; ++v) {
          bfpair p; p.u = *(const unsigned*)&VlT[(t*16 + lm)*LP + ks*32 + kbase(v, half)];
          vb[2*v] = p.b[0]; vb[2*v+1] = p.b[1];
        }
        cacc[t] = wmma_bf16(ka, vb, cacc[t]);
      }
    }
  }
#pragma unroll
  for (int t = 0; t < 3; ++t)
#pragma unroll
    for (int r = 0; r < 8; ++r)
      atomicAdd(&ctx[(bh*Mm + wv*16 + r + 8*half)*CE + t*16 + lm], cacc[t][r]);
}

// ---------------- kernel 4: o = D^-1 * (Qp @ ctx) ----------------------------
__global__ __launch_bounds__(256)
void k_out(const float* __restrict__ k0, const float* __restrict__ k1,
           const float* __restrict__ w,  const float* __restrict__ diag,
           const float* __restrict__ rowmax, const float* __restrict__ ctx,
           float* __restrict__ out0, float* __restrict__ out1) {
  __shared__ __bf16 swb[Mm*DK];        //  8 KB
  __shared__ __bf16 CbT[CE*LP];        // 12.4KB [col][m]
  __shared__ __bf16 Qt[8][16*LP];      // 33 KB  per-wave [l_local][m]
  int tid = threadIdx.x;
  for (int i = tid; i < Mm*DK; i += 256) swb[i] = f2bf(w[i]);
  int bh = blockIdx.x >> 6, l0 = (blockIdx.x & 63) * 128;
  int b = bh >> 3, h = bh & 7;
  // stage ctx slice transposed: CbT[col][m] = ctx[bh][m][col]
  for (int i = tid; i < Mm*CE; i += 256) {
    int m = i / CE, col = i % CE;
    CbT[col*LP + m] = f2bf(ctx[(bh*Mm + m)*CE + col]);
  }
  __syncthreads();
  int lane = tid & 31, wv = tid >> 5, lm = lane & 15, half = lane >> 4;
  v16bf wB[8]; buildWB(swb, lm, half, wB);

  // recompute dd for this wave's 16 rows, exp with per-row max -> Qp in LDS
  {
    int lrow = wv*16 + lm;
    long n = (long)b * Ll + l0 + lrow;
    v16bf a;
#pragma unroll
    for (int j = 0; j < 16; ++j)
      a[j] = f2bf(featld(k0, k1, n, h, kpat(j, half)) * DNc);
    float sub[8];
#pragma unroll
    for (int r = 0; r < 8; ++r) {
      int rr = bh*Ll + l0 + wv*16 + r + 8*half;
      sub[r] = diag[rr] + rowmax[rr];
    }
    v8f zero = {};
#pragma unroll
    for (int t = 0; t < 8; ++t) {
      v8f acc = wmma_bf16(a, wB[t], zero);
#pragma unroll
      for (int r = 0; r < 8; ++r)
        Qt[wv][(r + 8*half)*LP + t*16 + lm] =
            f2bf(RATIOc * (__expf(acc[r] - sub[r]) + EPSc));
    }
  }
  __syncthreads();
  // out GEMM: A = Qp [16 x 128], B = ctx [128 x 48] (col 32 gives D)
  v8f oacc[3] = {};
#pragma unroll
  for (int ks = 0; ks < 4; ++ks) {
    v16bf qa;
#pragma unroll
    for (int v = 0; v < 8; ++v) {
      bfpair p; p.u = *(const unsigned*)&Qt[wv][lm*LP + ks*32 + kbase(v, half)];
      qa[2*v] = p.b[0]; qa[2*v+1] = p.b[1];
    }
#pragma unroll
    for (int t = 0; t < 3; ++t) {
      v16bf cb;
#pragma unroll
      for (int v = 0; v < 8; ++v) {
        bfpair p; p.u = *(const unsigned*)&CbT[(t*16 + lm)*LP + ks*32 + kbase(v, half)];
        cb[2*v] = p.b[0]; cb[2*v+1] = p.b[1];
      }
      oacc[t] = wmma_bf16(qa, cb, oacc[t]);
    }
  }
  // scale by 1/D (col 32 lives in tile 2 at lane%16==0 of each half) and write
#pragma unroll
  for (int r = 0; r < 8; ++r) {
    float d = __shfl(oacc[2][r], lane & 16, 32);
    float dinv = 1.0f / d;
    int l = l0 + wv*16 + r + 8*half;
    long n2 = (long)b * Ll + l;
#pragma unroll
    for (int t = 0; t < 2; ++t) {
      int col = t*16 + lm;
      float val = oacc[t][r] * dinv;
      if (col < 8) out0[n2*64 + h*8 + col] = val;
      else { int idx = col - 8; out1[(n2*64 + h*8 + idx/3)*3 + idx%3] = val; }
    }
  }
}

// ---------------- launcher ---------------------------------------------------
extern "C" void kernel_launch(void* const* d_in, const int* in_sizes, int n_in,
                              void* d_out, int out_size, void* d_ws, size_t ws_size,
                              hipStream_t stream) {
  (void)in_sizes; (void)n_in; (void)out_size; (void)ws_size;
  const float* v0 = (const float*)d_in[0];
  const float* v1 = (const float*)d_in[1];
  const float* k0 = (const float*)d_in[2];
  const float* k1 = (const float*)d_in[3];
  // d_in[4], d_in[5] (q0,q1) are mathematically unused (K = Q from k-dict)
  const float* w  = (const float*)d_in[6];

  float* out0 = (float*)d_out;                 // [N,64,1]
  float* out1 = out0 + (long)Nn * 64;          // [N,64,3]

  float* diag   = (float*)d_ws;                // BHL
  float* rowmax = diag + BHL;                  // BHL
  float* gmax   = rowmax + BHL;                // 1 (padded)
  float* ctx    = gmax + 64;                   // BH*128*48

  k_stats<<<BHL/128, 256, 0, stream>>>(k0, k1, w, diag, rowmax);
  k_init <<<(BH*Mm*CE + 255)/256, 256, 0, stream>>>(ctx);
  k_gmax <<<1, 1024, 0, stream>>>(rowmax, gmax);
  k_ctx  <<<BH*8, 256, 0, stream>>>(k0, k1, v0, v1, w, diag, gmax, ctx);
  k_out  <<<BH*64, 256, 0, stream>>>(k0, k1, w, diag, rowmax, ctx, out0, out1);
}
